// SConv_55387898249896
// MI455X (gfx1250) — matrix operand order repaired and checked by reading, not compiled, not run on hardware
//
#include <hip/hip_runtime.h>
#include <hip/hip_bf16.h>

#define D        64
#define NKER     25
#define NNODES   40000
#define NEDGES   640000

typedef __attribute__((ext_vector_type(16))) __bf16 v16bf;
typedef __attribute__((ext_vector_type(8)))  float  v8f;

struct Pack32 { uint4 lo, hi; };   // 32 bytes == v16bf

static __device__ __forceinline__ unsigned short f2bf(float f) {
    unsigned u = __float_as_uint(f);
    u += 0x7FFFu + ((u >> 16) & 1u);          // round-to-nearest-even
    return (unsigned short)(u >> 16);
}
static __device__ __forceinline__ float bf2f(unsigned short h) {
    return __uint_as_float(((unsigned)h) << 16);
}

// native f32 max atomic (lowers to global_atomic_max_num_f32); agg pre-filled with -inf
static __device__ __forceinline__ void atomicMaxF32(float* addr, float v) {
    unsafeAtomicMax(addr, v);
}

// ---------------- prep kernels ----------------

__global__ void cvt_bf16_kernel(const float* __restrict__ x,
                                unsigned short* __restrict__ xb, int n) {
    int i = blockIdx.x * blockDim.x + threadIdx.x;
    if (i < n) xb[i] = f2bf(x[i]);
}

// Wt[k][o][i] = bf16(W[k][i][o])  (transpose so WMMA B fragments are contiguous)
__global__ void prep_w_kernel(const float* __restrict__ W,
                              unsigned short* __restrict__ Wt) {
    int t = blockIdx.x * blockDim.x + threadIdx.x;
    if (t >= NKER * D * D) return;
    int k = t / (D * D);
    int r = t % (D * D);
    int o = r / D;
    int i = r % D;
    Wt[t] = f2bf(W[k * D * D + i * D + o]);
}

__global__ void fill_neginf_kernel(float* __restrict__ p, int n) {
    int i = blockIdx.x * blockDim.x + threadIdx.x;
    if (i < n) p[i] = __uint_as_float(0xFF800000u);   // -inf
}

// ---------------- WMMA GEMM:  Yb[n][k][o] = bf16( sum_i xb[n][i] * W[k][i][o] ) -------------
// grid = (ceil(2500/8), 25), block = 256 (8 waves); 1 wave -> 16 nodes x 64 outs for kernel k

__global__ __launch_bounds__(256) void
spline_gemm_kernel(const unsigned short* __restrict__ xb,   // [N][64] bf16
                   const unsigned short* __restrict__ Wt,   // [25][64][64] bf16 ([k][o][i])
                   unsigned short* __restrict__ Yb)         // [N][25][64] bf16
{
    __shared__ unsigned short Wlds[D * D];          // 8 KB, W[k] tile
    __shared__ unsigned short outLds[8][16 * D];    // 16 KB, per-wave D repack

    const int k   = blockIdx.y;
    const int tid = threadIdx.x;

    // stage Wt[k] (4096 ushorts = 512 uint4) with 256 threads
    {
        const uint4* s = (const uint4*)(Wt + k * D * D);
        uint4*       d = (uint4*)Wlds;
        d[tid]       = s[tid];
        d[tid + 256] = s[tid + 256];
    }
    __syncthreads();

    const int wave    = tid >> 5;
    const int lane    = tid & 31;
    const int nt      = blockIdx.x * 8 + wave;      // 16-node tile index
    if (nt >= (NNODES / 16)) return;

    const int halfsel = lane >> 4;                  // 0: lanes 0-15, 1: lanes 16-31
    const int lm      = lane & 15;

    // A fragments (16x32 bf16 layout): lane half selects K-offset 0/8 within each 16-K group
    const unsigned short* xrow = xb + (nt * 16 + lm) * D;
    v16bf a[2];
#pragma unroll
    for (int s = 0; s < 2; ++s) {
        Pack32 p;
        const int off = s * 32 + halfsel * 8;
        p.lo = *(const uint4*)(xrow + off);         // K = off..off+7
        p.hi = *(const uint4*)(xrow + off + 16);    // K = off+16..off+23
        a[s] = __builtin_bit_cast(v16bf, p);
    }

    unsigned short* myOut = outLds[wave];
#pragma unroll
    for (int t = 0; t < 4; ++t) {                   // 4 N-tiles of 16 outputs
        v8f c = {};
#pragma unroll
        for (int s = 0; s < 2; ++s) {               // K = 0..31, 32..63
            Pack32 p;
            const unsigned short* bp = Wlds + (t * 16 + lm) * D + s * 32 + halfsel * 8;
            p.lo = *(const uint4*)(bp);             // K = base..base+7
            p.hi = *(const uint4*)(bp + 16);        // K = base+16..base+23
            v16bf bf = __builtin_bit_cast(v16bf, p);
            c = __builtin_amdgcn_wmma_f32_16x16x32_bf16(false, a[s], false, bf,
                                                        (short)0, c, false, false);
        }
        // D layout: VGPR r -> (M = r + 8*halfsel, N = lm); repack into [m][o] staging
#pragma unroll
        for (int r = 0; r < 8; ++r) {
            const int m = r + halfsel * 8;
            myOut[m * D + t * 16 + lm] = f2bf(c[r]);
        }
    }
    // wave-private LDS: enforce store->load ordering without a workgroup barrier
    asm volatile("s_wait_dscnt 0" ::: "memory");

    // coalesced store: 16 rows x 128B; 2 lanes per row, 4x uint4 each
    {
        const int m = lane >> 1;
        const int p = lane & 1;
        const long long base = (long long)(nt * 16 + m) * (NKER * D) + k * D;
        const uint4* sq = (const uint4*)(myOut + m * D);
        uint4*       dq = (uint4*)(Yb + base);
#pragma unroll
        for (int j = 0; j < 4; ++j)
            dq[p * 4 + j] = sq[p * 4 + j];
    }
}

// ---------------- edge kernel: gather 4 bf16 rows, weighted-sum, atomic segment-max ----------
// 8 threads per edge; thread owns 8 contiguous outputs -> one uint4 (16B) gather per tap

__global__ __launch_bounds__(256) void
spline_edge_kernel(const long long* __restrict__ ei,        // [2][E] int64
                   const float* __restrict__ ea,            // [E][2]
                   const unsigned short* __restrict__ Yb,   // [N][25][64] bf16
                   float* __restrict__ agg)                 // [N][64] f32 (init -inf)
{
    const int t   = threadIdx.x;
    const int e   = blockIdx.x * 32 + (t >> 3);             // 32 edges per block
    if (e >= NEDGES) return;
    const int sub = t & 7;                                  // outputs o = sub*8 .. sub*8+7

    const int src = (int)ei[e];
    const int dst = (int)ei[NEDGES + e];

    // degree-1 open B-spline basis, K=5
    const float v0 = ea[e * 2 + 0] * 4.0f;
    const float v1 = ea[e * 2 + 1] * 4.0f;
    float b0 = fminf(fmaxf(floorf(v0), 0.0f), 3.0f);
    float b1 = fminf(fmaxf(floorf(v1), 0.0f), 3.0f);
    const float f0 = v0 - b0, f1 = v1 - b1;
    const int i0 = (int)b0, i1 = (int)b1;

    float w[4];
    int   idx[4];
    w[0] = (1.f - f0) * (1.f - f1); idx[0] = i0 * 5 + i1;
    w[1] = (1.f - f0) * f1;         idx[1] = i0 * 5 + i1 + 1;
    w[2] = f0 * (1.f - f1);         idx[2] = (i0 + 1) * 5 + i1;
    w[3] = f0 * f1;                 idx[3] = (i0 + 1) * 5 + i1 + 1;

    float acc[8];
#pragma unroll
    for (int j = 0; j < 8; ++j) acc[j] = 0.0f;

    const long long srcBase = (long long)src * (NKER * D) + sub * 8;
#pragma unroll
    for (int s = 0; s < 4; ++s) {
        const uint4 q = *(const uint4*)(Yb + srcBase + idx[s] * D);   // 8 bf16
        const unsigned u0 = q.x, u1 = q.y, u2 = q.z, u3 = q.w;
        acc[0] += w[s] * bf2f((unsigned short)(u0 & 0xFFFFu));
        acc[1] += w[s] * bf2f((unsigned short)(u0 >> 16));
        acc[2] += w[s] * bf2f((unsigned short)(u1 & 0xFFFFu));
        acc[3] += w[s] * bf2f((unsigned short)(u1 >> 16));
        acc[4] += w[s] * bf2f((unsigned short)(u2 & 0xFFFFu));
        acc[5] += w[s] * bf2f((unsigned short)(u2 >> 16));
        acc[6] += w[s] * bf2f((unsigned short)(u3 & 0xFFFFu));
        acc[7] += w[s] * bf2f((unsigned short)(u3 >> 16));
    }

    float* ap = agg + (long long)dst * D + sub * 8;
#pragma unroll
    for (int j = 0; j < 8; ++j)
        atomicMaxF32(ap + j, acc[j]);
}

// ---------------- combine: out = fix(agg) + x@root + bias, optional relu + bf16 copy --------

__global__ __launch_bounds__(256) void
spline_combine_kernel(const float* __restrict__ xin,        // [N][64] layer input
                      const float* __restrict__ agg,        // [N][64]
                      const float* __restrict__ root,       // [64][64]
                      const float* __restrict__ bias,       // [64]
                      float* __restrict__ out,              // [N][64]
                      unsigned short* __restrict__ outb,    // bf16 copy (or nullptr)
                      int do_relu)
{
    __shared__ float xs[4 * D];
    const int tid = threadIdx.x;
    xs[tid] = xin[(long long)blockIdx.x * 4 * D + tid];     // 4 nodes per block
    __syncthreads();

    const int local = tid >> 6;
    const int o     = tid & 63;
    const long long node = (long long)blockIdx.x * 4 + local;

    float acc = bias[o];
    const float* xr = xs + local * D;
#pragma unroll 8
    for (int i = 0; i < D; ++i)
        acc += xr[i] * root[i * D + o];

    float a = agg[node * D + o];
    if (!__builtin_isfinite(a)) a = 0.0f;           // no-incoming-edge nodes
    float r = acc + a;
    if (do_relu) r = fmaxf(r, 0.0f);
    out[node * D + o] = r;
    if (outb) outb[node * D + o] = f2bf(r);
}

// ---------------- launch ----------------

extern "C" void kernel_launch(void* const* d_in, const int* in_sizes, int n_in,
                              void* d_out, int out_size, void* d_ws, size_t ws_size,
                              hipStream_t stream) {
    (void)in_sizes; (void)n_in; (void)out_size; (void)ws_size;
    const float*     x     = (const float*)d_in[0];
    const long long* ei    = (const long long*)d_in[1];
    const float*     ea    = (const float*)d_in[2];
    const float*     W1    = (const float*)d_in[3];
    const float*     root1 = (const float*)d_in[4];
    const float*     bias1 = (const float*)d_in[5];
    const float*     W2    = (const float*)d_in[6];
    const float*     root2 = (const float*)d_in[7];
    const float*     bias2 = (const float*)d_in[8];
    float*           out   = (float*)d_out;

    char* ws = (char*)d_ws;
    size_t off = 0;
    auto alloc = [&](size_t bytes) -> char* {
        char* p = ws + off;
        off = (off + bytes + 255) & ~(size_t)255;
        return p;
    };
    unsigned short* Wt1 = (unsigned short*)alloc((size_t)NKER * D * D * 2);
    unsigned short* Wt2 = (unsigned short*)alloc((size_t)NKER * D * D * 2);
    unsigned short* xb  = (unsigned short*)alloc((size_t)NNODES * D * 2);
    unsigned short* hb  = (unsigned short*)alloc((size_t)NNODES * D * 2);
    float*          h   = (float*)alloc((size_t)NNODES * D * 4);
    float*          agg = (float*)alloc((size_t)NNODES * D * 4);
    unsigned short* Yb  = (unsigned short*)alloc((size_t)NNODES * NKER * D * 2); // 128 MB, fits L2

    const int nElem = NNODES * D;
    prep_w_kernel<<<(NKER * D * D + 255) / 256, 256, 0, stream>>>(W1, Wt1);
    prep_w_kernel<<<(NKER * D * D + 255) / 256, 256, 0, stream>>>(W2, Wt2);
    cvt_bf16_kernel<<<(nElem + 255) / 256, 256, 0, stream>>>(x, xb, nElem);

    dim3 ggrid((NNODES / 16 + 7) / 8, NKER);

    // ---- layer 1 ----
    fill_neginf_kernel<<<(nElem + 255) / 256, 256, 0, stream>>>(agg, nElem);
    spline_gemm_kernel<<<ggrid, 256, 0, stream>>>(xb, Wt1, Yb);
    spline_edge_kernel<<<NEDGES / 32, 256, 0, stream>>>(ei, ea, Yb, agg);
    spline_combine_kernel<<<NNODES / 4, 256, 0, stream>>>(x, agg, root1, bias1, h, hb, 1);

    // ---- layer 2 ----
    fill_neginf_kernel<<<(nElem + 255) / 256, 256, 0, stream>>>(agg, nElem);
    spline_gemm_kernel<<<ggrid, 256, 0, stream>>>(hb, Wt2, Yb);
    spline_edge_kernel<<<NEDGES / 32, 256, 0, stream>>>(ei, ea, Yb, agg);
    spline_combine_kernel<<<NNODES / 4, 256, 0, stream>>>(h, agg, root2, bias2, out,
                                                          (unsigned short*)nullptr, 0);
}
